// ModelMarkovVaeCtc_23029614641826
// MI455X (gfx1250) — matrix operand order repaired and checked by reading, not compiled, not run on hardware
//
#include <hip/hip_runtime.h>
#include <math.h>

// ---------------------------------------------------------------------------
// MI455X (gfx1250) implementation.
//   - bf16 WMMA 16x16x32, f32 accumulate, for all GEMMs.
//   - One-time prep kernels pack fp32 weights -> bf16 WMMA B-fragments in d_ws
//     and X -> bf16 padded [64000][96].
//   - fused_parallel_kernel: q-nets + reparam + decoder (4000 blocks x 4 waves).
//   - markov_scan_kernel: sequential prior; 2 independent batch-half chains,
//     16 waves each, activations in LDS, 2000 steps.
// Workspace use: ~14.3 MB of d_ws.
// ---------------------------------------------------------------------------

typedef __attribute__((ext_vector_type(16))) __bf16 v16bf;
typedef __attribute__((ext_vector_type(8)))  float  v8f;

#define DEV static __device__ __forceinline__

DEV __bf16 f2bf(float f) {
  unsigned u = __builtin_bit_cast(unsigned, f);
  u += 0x7fffu + ((u >> 16) & 1u);               // round-to-nearest-even
  unsigned short h = (unsigned short)(u >> 16);
  return __builtin_bit_cast(__bf16, h);
}
DEV float bf2f(__bf16 b) {
  unsigned short h = __builtin_bit_cast(unsigned short, b);
  unsigned u = ((unsigned)h) << 16;
  return __builtin_bit_cast(float, u);
}
DEV v8f vzero() { v8f z = {0.f,0.f,0.f,0.f,0.f,0.f,0.f,0.f}; return z; }

DEV v8f wmma_bf16(v16bf a, v16bf b, v8f c) {
  return __builtin_amdgcn_wmma_f32_16x16x32_bf16(false, a, false, b, (short)0, c,
                                                 false, false);
}

// A fragment (16x32 bf16, M x K) gathered from row-major LDS tile.
// Lane l: m = l%16, h = l/16.  Element pair v (VGPR v), slot s:
//   K = h*16 + v*2 + s       (contiguous, matches packed B)
DEV v16bf load_a_frag(const __bf16* base, int lda, int lane, int k0) {
  const int m = lane & 15;
  const int h = lane >> 4;
  const __bf16* row = base + m * lda + k0 + h * 16;
  v16bf a;
#pragma unroll
  for (int v = 0; v < 8; ++v) {
    const int kk = v << 1;  // {0,2,4,...,14}
    a[2 * v]     = row[kk];
    a[2 * v + 1] = row[kk + 1];
  }
  return a;
}

// B fragments are pre-packed lane-contiguous: frag (nt,kt): lane l holds
// 16 bf16 = column n = nt*16 + l%16, K = kt*32 + (l/16)*16 + j.
template <int NT>
DEV void gemm_acc(const __bf16* A, int lda, const __bf16* Bp, int ktiles,
                  int nt0, int lane, v8f* acc) {
  for (int kt = 0; kt < ktiles; ++kt) {
    v16bf a = load_a_frag(A, lda, lane, kt * 32);
#pragma unroll
    for (int i = 0; i < NT; ++i) {
      const __bf16* p = Bp + ((((size_t)(nt0 + i)) * ktiles + kt) * 32 + lane) * 16;
      v16bf b = *(const v16bf*)p;
      acc[i] = wmma_bf16(a, b, acc[i]);
    }
  }
}

// D layout: lane l holds column n = l%16, rows m = r + 8*(l/16), r=0..7.
DEV void relu_store_lds(__bf16* dst, int ldn, int lane, int n0,
                        const float* bias, v8f acc) {
  const int n = n0 + (lane & 15);
  const int mh = (lane >> 4) * 8;
  const float b = bias[n];
#pragma unroll
  for (int r = 0; r < 8; ++r) {
    float v = acc[r] + b;
    dst[(mh + r) * ldn + n] = f2bf(v > 0.f ? v : 0.f);
  }
}

// ------------------------------- prep kernels ------------------------------

__global__ void pack_weight_kernel(const float* W, __bf16* out, int K, int N, int Kp) {
  const int ktiles = Kp >> 5;
  const int ntiles = N >> 4;
  const size_t total = (size_t)ntiles * ktiles * 512;
  for (size_t idx = (size_t)blockIdx.x * blockDim.x + threadIdx.x; idx < total;
       idx += (size_t)gridDim.x * blockDim.x) {
    const int j    = (int)(idx & 15);
    const int lane = (int)((idx >> 4) & 31);
    const size_t f = idx >> 9;
    const int kt = (int)(f % ktiles);
    const int nt = (int)(f / ktiles);
    const int k  = kt * 32 + (lane >> 4) * 16 + j;
    const int nn = nt * 16 + (lane & 15);
    out[idx] = f2bf((k < K) ? W[(size_t)k * N + nn] : 0.f);
  }
}

__global__ void convert_x_kernel(const float* X, __bf16* out, long long rows) {
  const long long total = rows * 96;
  for (long long idx = (long long)blockIdx.x * blockDim.x + threadIdx.x; idx < total;
       idx += (long long)gridDim.x * blockDim.x) {
    const long long r = idx / 96;
    const int c = (int)(idx % 96);
    out[idx] = (c < 80) ? f2bf(X[r * 80 + c]) : f2bf(0.f);
  }
}

// ------------------------- parallel part (q + decoder) ---------------------

struct PArgs {
  const float* X;
  const float* eps;
  const float *qmu_b1, *qmu_b2, *qsg_b1, *qsg_b2, *lz_b, *hid_b, *out_b;
  const __bf16 *qmuW1, *qmuW2, *qsgW1, *qsgW2, *lzW, *hidWx, *hidWz, *outW;
  float *o_logits, *o_muq, *o_sgq;
};

__global__ __launch_bounds__(128) void fused_parallel_kernel(PArgs P) {
  __shared__ __bf16 sX[16 * 96];    // X tile, bf16, K padded 80->96
  __shared__ __bf16 sA[16 * 256];   // activation buffer 1
  __shared__ __bf16 sB[16 * 256];   // activation buffer 2

  const int tid  = threadIdx.x;
  const int lane = tid & 31;
  const int wave = tid >> 5;        // 4 waves, each owns 4 N-subtiles
  const size_t row0 = (size_t)blockIdx.x * 16;
  const int nt0  = wave * 4;
  const int nloc = lane & 15;
  const int mh   = (lane >> 4) * 8;

  for (int i = tid; i < 16 * 96; i += 128) {
    const int r = i / 96, c = i % 96;
    sX[i] = (c < 80) ? f2bf(P.X[(row0 + r) * 80 + c]) : f2bf(0.f);
  }
  __syncthreads();

  v8f acc[4];

  // ---- q_mu layer 1: h = relu(X @ W1 + b1) -> sA
  for (int i = 0; i < 4; ++i) acc[i] = vzero();
  gemm_acc<4>(sX, 96, P.qmuW1, 3, nt0, lane, acc);
  for (int i = 0; i < 4; ++i)
    relu_store_lds(sA, 256, lane, (nt0 + i) * 16, P.qmu_b1, acc[i]);
  __syncthreads();

  // ---- mu_q = h @ W2 + b2 -> global (f32) + sB (bf16)
  for (int i = 0; i < 4; ++i) acc[i] = vzero();
  gemm_acc<4>(sA, 256, P.qmuW2, 8, nt0, lane, acc);
  for (int i = 0; i < 4; ++i) {
    const int n = (nt0 + i) * 16 + nloc;
    const float b = P.qmu_b2[n];
#pragma unroll
    for (int r = 0; r < 8; ++r) {
      const float v = acc[i][r] + b;
      P.o_muq[(row0 + mh + r) * 256 + n] = v;
      sB[(mh + r) * 256 + n] = f2bf(v);
    }
  }
  __syncthreads();

  // ---- q_sigma layer 1 -> sA (overwrite)
  for (int i = 0; i < 4; ++i) acc[i] = vzero();
  gemm_acc<4>(sX, 96, P.qsgW1, 3, nt0, lane, acc);
  for (int i = 0; i < 4; ++i)
    relu_store_lds(sA, 256, lane, (nt0 + i) * 16, P.qsg_b1, acc[i]);
  __syncthreads();

  // ---- sigma_q = h @ W2 + b2; Z = mu + exp(sigma/2)*eps
  for (int i = 0; i < 4; ++i) acc[i] = vzero();
  gemm_acc<4>(sA, 256, P.qsgW2, 8, nt0, lane, acc);
  float zv[4][8];
  for (int i = 0; i < 4; ++i) {
    const int n = (nt0 + i) * 16 + nloc;
    const float b = P.qsg_b2[n];
    const float e = P.eps[n];
#pragma unroll
    for (int r = 0; r < 8; ++r) {
      const float s = acc[i][r] + b;
      P.o_sgq[(row0 + mh + r) * 256 + n] = s;
      zv[i][r] = bf2f(sB[(mh + r) * 256 + n]) + __expf(s * 0.5f) * e;
    }
  }
  __syncthreads();                                 // all reads of sA done
  for (int i = 0; i < 4; ++i) {
    const int n = (nt0 + i) * 16 + nloc;
#pragma unroll
    for (int r = 0; r < 8; ++r) sA[(mh + r) * 256 + n] = f2bf(zv[i][r]);  // Z
  }
  __syncthreads();

  // ---- tilde_Z = relu(Z @ Wz + bz) -> sB
  for (int i = 0; i < 4; ++i) acc[i] = vzero();
  gemm_acc<4>(sA, 256, P.lzW, 8, nt0, lane, acc);
  for (int i = 0; i < 4; ++i)
    relu_store_lds(sB, 256, lane, (nt0 + i) * 16, P.lz_b, acc[i]);
  __syncthreads();

  // ---- hidden: h = relu(X@Wx + tz@Wz2 + bh) -> sA
  for (int i = 0; i < 4; ++i) acc[i] = vzero();
  gemm_acc<4>(sX, 96, P.hidWx, 3, nt0, lane, acc);
  gemm_acc<4>(sB, 256, P.hidWz, 8, nt0, lane, acc);
  for (int i = 0; i < 4; ++i)
    relu_store_lds(sA, 256, lane, (nt0 + i) * 16, P.hid_b, acc[i]);
  __syncthreads();

  // ---- logits = h @ Wo + bo (N = 32 -> waves 0,1 only)
  if (wave < 2) {
    v8f oc = vzero();
    gemm_acc<1>(sA, 256, P.outW, 8, wave, lane, &oc);
    const int n = wave * 16 + nloc;
    const float b = P.out_b[n];
#pragma unroll
    for (int r = 0; r < 8; ++r)
      P.o_logits[(row0 + mh + r) * 32 + n] = oc[r] + b;
  }
}

// ---------------------------- sequential prior -----------------------------

struct SArgs {
  const __bf16* Xbf;
  const float *b1mu0, *b2mu0, *b1sg0, *b2sg0, *b1mu, *b2mu, *b1sg, *b2sg;
  const __bf16 *W1mu0, *W2mu0, *W1sg0, *W2sg0, *W1mu, *W2mu, *W1sg, *W2sg;
  float *o_mup, *o_sgp;
};

__global__ __launch_bounds__(512) void markov_scan_kernel(SArgs S) {
  // tilde layout per row: [0..255]=mu_prev [256..511]=sig_prev [512..607]=x_t(pad)
  __shared__ __bf16 sT[16 * 608];
  __shared__ __bf16 sH[16 * 256];

  const int tid  = threadIdx.x;
  const int lane = tid & 31;
  const int nt   = tid >> 5;           // 16 waves; wave == N-subtile
  const int b0   = blockIdx.x * 16;    // independent batch half
  const int nloc = lane & 15;
  const int n    = nt * 16 + nloc;
  const int mh   = (lane >> 4) * 8;

  for (int t = 0; t < 2000; ++t) {
    // stage x_t (bf16, already padded to 96 with zeros)
    for (int i = tid; i < 16 * 96; i += 512) {
      const int r = i / 96, c = i % 96;
      sT[r * 608 + 512 + c] = S.Xbf[((size_t)(b0 + r) * 2000 + t) * 96 + c];
    }
    __syncthreads();  // B1

    const bool first = (t == 0);
    const __bf16* Abase = first ? (sT + 512) : sT;
    const int kt1 = first ? 3 : 19;                    // K = 96 or 608
    const __bf16* W1m = first ? S.W1mu0 : S.W1mu;
    const __bf16* W2m = first ? S.W2mu0 : S.W2mu;
    const __bf16* W1s = first ? S.W1sg0 : S.W1sg;
    const __bf16* W2s = first ? S.W2sg0 : S.W2sg;
    const float* c1m = first ? S.b1mu0 : S.b1mu;
    const float* c2m = first ? S.b2mu0 : S.b2mu;
    const float* c1s = first ? S.b1sg0 : S.b1sg;
    const float* c2s = first ? S.b2sg0 : S.b2sg;

    // mu layer 1
    v8f h = vzero();
    gemm_acc<1>(Abase, 608, W1m, kt1, nt, lane, &h);
    relu_store_lds(sH, 256, lane, nt * 16, c1m, h);
    __syncthreads();  // B2

    // mu layer 2
    v8f mu = vzero();
    gemm_acc<1>(sH, 256, W2m, 8, nt, lane, &mu);
    {
      const float b = c2m[n];
#pragma unroll
      for (int r = 0; r < 8; ++r) {
        mu[r] += b;
        S.o_mup[((size_t)(b0 + mh + r) * 2000 + t) * 256 + n] = mu[r];
      }
    }
    __syncthreads();  // B3: sH reads done before overwrite

    // sigma layer 1
    h = vzero();
    gemm_acc<1>(Abase, 608, W1s, kt1, nt, lane, &h);
    relu_store_lds(sH, 256, lane, nt * 16, c1s, h);
    __syncthreads();  // B4: also guarantees all sT reads of this step are done

    // sigma layer 2
    v8f sg = vzero();
    gemm_acc<1>(sH, 256, W2s, 8, nt, lane, &sg);
    {
      const float b = c2s[n];
#pragma unroll
      for (int r = 0; r < 8; ++r) {
        sg[r] += b;
        S.o_sgp[((size_t)(b0 + mh + r) * 2000 + t) * 256 + n] = sg[r];
      }
    }
    // publish carry for next step (wave-exclusive columns; B1 orders reads)
#pragma unroll
    for (int r = 0; r < 8; ++r) {
      sT[(mh + r) * 608 + n]       = f2bf(mu[r]);
      sT[(mh + r) * 608 + 256 + n] = f2bf(sg[r]);
    }
  }
}

// ------------------------------- launcher ----------------------------------

extern "C" void kernel_launch(void* const* d_in, const int* in_sizes, int n_in,
                              void* d_out, int out_size, void* d_ws, size_t ws_size,
                              hipStream_t stream) {
  (void)in_sizes; (void)n_in; (void)out_size; (void)ws_size;
  // setup_inputs() dict (insertion) order, params leaves in declaration order.
  const float* X       = (const float*)d_in[0];
  const float* eps     = (const float*)d_in[1];
  const float* pmu0_W1 = (const float*)d_in[2];
  const float* pmu0_b1 = (const float*)d_in[3];
  const float* pmu0_W2 = (const float*)d_in[4];
  const float* pmu0_b2 = (const float*)d_in[5];
  const float* psg0_W1 = (const float*)d_in[6];
  const float* psg0_b1 = (const float*)d_in[7];
  const float* psg0_W2 = (const float*)d_in[8];
  const float* psg0_b2 = (const float*)d_in[9];
  const float* pmu_W1  = (const float*)d_in[10];
  const float* pmu_b1  = (const float*)d_in[11];
  const float* pmu_W2  = (const float*)d_in[12];
  const float* pmu_b2  = (const float*)d_in[13];
  const float* psg_W1  = (const float*)d_in[14];
  const float* psg_b1  = (const float*)d_in[15];
  const float* psg_W2  = (const float*)d_in[16];
  const float* psg_b2  = (const float*)d_in[17];
  const float* qmu_W1  = (const float*)d_in[18];
  const float* qmu_b1  = (const float*)d_in[19];
  const float* qmu_W2  = (const float*)d_in[20];
  const float* qmu_b2  = (const float*)d_in[21];
  const float* qsg_W1  = (const float*)d_in[22];
  const float* qsg_b1  = (const float*)d_in[23];
  const float* qsg_W2  = (const float*)d_in[24];
  const float* qsg_b2  = (const float*)d_in[25];
  const float* lz_W    = (const float*)d_in[26];
  const float* lz_b    = (const float*)d_in[27];
  const float* hid_W   = (const float*)d_in[28];
  const float* hid_b   = (const float*)d_in[29];
  const float* out_W   = (const float*)d_in[30];
  const float* out_b   = (const float*)d_in[31];

  __bf16* ws = (__bf16*)d_ws;
  size_t off = 0;
  auto take = [&](size_t ne) { __bf16* p = ws + off; off += ne; return p; };

  __bf16* Xbf     = take(64000ull * 96);
  __bf16* P_qmuW1 = take(96 * 256);
  __bf16* P_qmuW2 = take(256 * 256);
  __bf16* P_qsgW1 = take(96 * 256);
  __bf16* P_qsgW2 = take(256 * 256);
  __bf16* P_lzW   = take(256 * 256);
  __bf16* P_hidWx = take(96 * 256);
  __bf16* P_hidWz = take(256 * 256);
  __bf16* P_outW  = take(256 * 32);
  __bf16* P_mu0W1 = take(96 * 256);
  __bf16* P_mu0W2 = take(256 * 256);
  __bf16* P_sg0W1 = take(96 * 256);
  __bf16* P_sg0W2 = take(256 * 256);
  __bf16* P_muW1  = take(608 * 256);
  __bf16* P_muW2  = take(256 * 256);
  __bf16* P_sgW1  = take(608 * 256);
  __bf16* P_sgW2  = take(256 * 256);

  convert_x_kernel<<<2048, 256, 0, stream>>>(X, Xbf, 64000);
  pack_weight_kernel<<<96,  256, 0, stream>>>(qmu_W1, P_qmuW1, 80, 256, 96);
  pack_weight_kernel<<<256, 256, 0, stream>>>(qmu_W2, P_qmuW2, 256, 256, 256);
  pack_weight_kernel<<<96,  256, 0, stream>>>(qsg_W1, P_qsgW1, 80, 256, 96);
  pack_weight_kernel<<<256, 256, 0, stream>>>(qsg_W2, P_qsgW2, 256, 256, 256);
  pack_weight_kernel<<<256, 256, 0, stream>>>(lz_W,   P_lzW,   256, 256, 256);
  pack_weight_kernel<<<96,  256, 0, stream>>>(hid_W,  P_hidWx, 80, 256, 96);
  pack_weight_kernel<<<256, 256, 0, stream>>>(hid_W + 80 * 256, P_hidWz, 256, 256, 256);
  pack_weight_kernel<<<32,  256, 0, stream>>>(out_W,  P_outW,  256, 32, 256);
  pack_weight_kernel<<<96,  256, 0, stream>>>(pmu0_W1, P_mu0W1, 80, 256, 96);
  pack_weight_kernel<<<256, 256, 0, stream>>>(pmu0_W2, P_mu0W2, 256, 256, 256);
  pack_weight_kernel<<<96,  256, 0, stream>>>(psg0_W1, P_sg0W1, 80, 256, 96);
  pack_weight_kernel<<<256, 256, 0, stream>>>(psg0_W2, P_sg0W2, 256, 256, 256);
  pack_weight_kernel<<<608, 256, 0, stream>>>(pmu_W1, P_muW1, 592, 256, 608);
  pack_weight_kernel<<<256, 256, 0, stream>>>(pmu_W2, P_muW2, 256, 256, 256);
  pack_weight_kernel<<<608, 256, 0, stream>>>(psg_W1, P_sgW1, 592, 256, 608);
  pack_weight_kernel<<<256, 256, 0, stream>>>(psg_W2, P_sgW2, 256, 256, 256);

  // d_out tuple layout (flat, return order): logits, mu_p, sigma_p, mu_q, sigma_q
  float* out = (float*)d_out;
  float* o_logits = out;                 // 32*2000*32
  float* o_mup = out + 2048000ll;        // 32*2000*256
  float* o_sgp = out + 18432000ll;
  float* o_muq = out + 34816000ll;
  float* o_sgq = out + 51200000ll;

  PArgs pa = { X, eps, qmu_b1, qmu_b2, qsg_b1, qsg_b2, lz_b, hid_b, out_b,
               P_qmuW1, P_qmuW2, P_qsgW1, P_qsgW2, P_lzW, P_hidWx, P_hidWz, P_outW,
               o_logits, o_muq, o_sgq };
  fused_parallel_kernel<<<4000, 128, 0, stream>>>(pa);

  SArgs sa = { Xbf,
               pmu0_b1, pmu0_b2, psg0_b1, psg0_b2, pmu_b1, pmu_b2, psg_b1, psg_b2,
               P_mu0W1, P_mu0W2, P_sg0W1, P_sg0W2, P_muW1, P_muW2, P_sgW1, P_sgW2,
               o_mup, o_sgp };
  markov_scan_kernel<<<2, 512, 0, stream>>>(sa);
}
